// Attention_31104153158141
// MI455X (gfx1250) — matrix operand order repaired and checked by reading, not compiled, not run on hardware
//
#include <hip/hip_runtime.h>
#include <hip/hip_bf16.h>

typedef __attribute__((ext_vector_type(16))) __bf16 v16bf;
typedef __attribute__((ext_vector_type(8)))  float  v8f;
typedef __attribute__((ext_vector_type(8)))  __bf16 bf8;
typedef __attribute__((ext_vector_type(4)))  __bf16 bf4;

static constexpr int Bn = 2;
static constexpr int Cc = 64;
static constexpr int Hh = 128;
static constexpr int Ww = 128;
static constexpr int Kk = 5;
static constexpr int TAPS = Kk * Kk;     // 25
static constexpr int SPAN = 48;          // padded col span staged per block
static constexpr int KP   = 160;         // attn-V GEMM K dim: 5 rows * 32 cols

__device__ __forceinline__ int iclamp(int x, int lo, int hi) {
    return x < lo ? lo : (x > hi ? hi : x);
}

__device__ __forceinline__ v16bf combine8(bf8 lo, bf8 hi) {
    return __builtin_shufflevector(lo, hi, 0, 1, 2, 3, 4, 5, 6, 7,
                                           8, 9, 10, 11, 12, 13, 14, 15);
}

// ---- CDNA5 async global->LDS copy (16B per lane), tracked on ASYNCcnt ----
__device__ __forceinline__ void async_copy_b128(const void* gsrc, void* ldst) {
    unsigned           dst = (unsigned)(uintptr_t)ldst;            // LDS byte address (low 32 bits)
    unsigned long long src = (unsigned long long)(uintptr_t)gsrc;  // 64-bit global address
    asm volatile("global_load_async_to_lds_b128 %0, %1, off"
                 :: "v"(dst), "v"(src) : "memory");
}
__device__ __forceinline__ void async_wait0() {
#if __has_builtin(__builtin_amdgcn_s_wait_asynccnt)
    __builtin_amdgcn_s_wait_asynccnt(0);
#else
    asm volatile("s_wait_asynccnt 0x0" ::: "memory");
#endif
}

// ---- WMMA fragment loaders (wave32, 16x16x32 bf16, contiguous k = 16*(lane/16) + s in both A and B) ----

__device__ __forceinline__ v16bf load_a_frag(const __bf16* s, int m0, int k0, int lda, int lane) {
    int m  = m0 + (lane & 15);
    int kb = (lane >> 4) << 4;
    const __bf16* p = s + m * lda + k0 + kb;
    return combine8(*(const bf8*)p, *(const bf8*)(p + 8));
}

__device__ __forceinline__ v16bf load_b_run(const __bf16* p) {
    const bf8* q = (const bf8*)p;
    return combine8(q[0], q[1]);
}

__device__ __forceinline__ v16bf load_b_frag_sw(const __bf16* s, int nt, int kblk, int lane) {
    return load_b_run(s + (((nt * 2 + kblk) * 32 + lane) << 4));
}

template <int N>
__device__ __forceinline__ void stage_weights_sw(__bf16* sW, const float* __restrict__ W, int tid) {
    constexpr int NT = N / 16;
    for (int e = tid; e < NT * 2 * 32 * 16; e += 256) {
        int j    = e & 15;
        int lane = (e >> 4) & 31;
        int kblk = (e >> 9) & 1;
        int nt   = e >> 10;
        int k = kblk * 32 + ((lane >> 4) << 4) + j;
        int n = nt * 16 + (lane & 15);
        sW[e] = (__bf16)W[k * N + n];
    }
}

// ------------------------------------------------------------------
// Stage 1: fused transpose + Q/KV projections (bf16 WMMA)
// ------------------------------------------------------------------
__global__ __launch_bounds__(256) void natten_proj_kernel(
    const float* __restrict__ q, const float* __restrict__ kv,
    const float* __restrict__ Wq, const float* __restrict__ bq,
    const float* __restrict__ Wkv, const float* __restrict__ bkv,
    __bf16* __restrict__ qq, __bf16* __restrict__ kk, __bf16* __restrict__ vv)
{
    __shared__ __align__(16) __bf16 sXq [Ww * Cc];
    __shared__ __align__(16) __bf16 sXkv[Ww * Cc];
    __shared__ __align__(16) __bf16 sWq [Cc * Cc];
    __shared__ __align__(16) __bf16 sWkv[Cc * 2 * Cc];

    const int tid = threadIdx.x;
    const int b = blockIdx.x / Hh;
    const int h = blockIdx.x % Hh;

    const float* qbase  = q  + ((size_t)b * Cc * Hh + h) * Ww;
    const float* kvbase = kv + ((size_t)b * Cc * Hh + h) * Ww;

    for (int e = tid; e < (Cc / 4) * Ww; e += 256) {
        int w  = e & (Ww - 1);
        int c4 = (e >> 7) << 2;
        bf4 xq, xk;
#pragma unroll
        for (int j = 0; j < 4; ++j) {
            xq[j] = (__bf16)qbase [(size_t)(c4 + j) * Hh * Ww + w];
            xk[j] = (__bf16)kvbase[(size_t)(c4 + j) * Hh * Ww + w];
        }
        *(bf4*)(sXq  + w * Cc + c4) = xq;
        *(bf4*)(sXkv + w * Cc + c4) = xk;
    }
    stage_weights_sw<Cc>(sWq, Wq, tid);
    stage_weights_sw<2 * Cc>(sWkv, Wkv, tid);
    __syncthreads();

    const int lane = tid & 31;
    const int m0 = (tid >> 5) * 16;

#pragma unroll
    for (int nt = 0; nt < 4; ++nt) {
        v8f acc = {};
#pragma unroll
        for (int kblk = 0; kblk < 2; ++kblk) {
            acc = __builtin_amdgcn_wmma_f32_16x16x32_bf16(
                false, load_a_frag(sXq, m0, kblk * 32, Cc, lane),
                false, load_b_frag_sw(sWq, nt, kblk, lane),
                (short)0, acc, false, false);
        }
        int n = nt * 16 + (lane & 15);
        float bn = bq[n];
#pragma unroll
        for (int r = 0; r < 8; ++r) {
            int m = m0 + r + ((lane >> 4) << 3);
            qq[(((size_t)b * Hh + h) * Ww + m) * Cc + n] = (__bf16)((acc[r] + bn) * 0.125f);
        }
    }

#pragma unroll
    for (int nt = 0; nt < 8; ++nt) {
        v8f acc = {};
#pragma unroll
        for (int kblk = 0; kblk < 2; ++kblk) {
            acc = __builtin_amdgcn_wmma_f32_16x16x32_bf16(
                false, load_a_frag(sXkv, m0, kblk * 32, Cc, lane),
                false, load_b_frag_sw(sWkv, nt, kblk, lane),
                (short)0, acc, false, false);
        }
        int n = nt * 16 + (lane & 15);
        float bn = bkv[n];
        __bf16* dst = (n < Cc) ? kk : vv;
        int nc = n & (Cc - 1);
#pragma unroll
        for (int r = 0; r < 8; ++r) {
            int m = m0 + r + ((lane >> 4) << 3);
            dst[(((size_t)b * Hh + h) * Ww + m) * Cc + nc] = (__bf16)(acc[r] + bn);
        }
    }
}

// ------------------------------------------------------------------
// Stage 2: windowed attention, fully on the matrix pipe.
//   scores: S[16p][5r*32cc] = Q(16x64) @ K^T   (K slab is already B-frag order)
//   attn-V: ctx[16p][64c]   = A(16x160) @ Vt   (A = scattered softmax weights)
// K/Q staging uses global_load_async_to_lds_b128 (ASYNCcnt).
// ------------------------------------------------------------------
__global__ __launch_bounds__(256) void natten_attn_kernel(
    const __bf16* __restrict__ qq, const __bf16* __restrict__ kk,
    const __bf16* __restrict__ vv, const float* __restrict__ rpb,
    __bf16* __restrict__ ctx)
{
    // phase A: sK[5][SPAN][64] + sQ[32][64]   |   phase B: sVt[2][5][64][32]
    __shared__ __align__(16) unsigned char uBuf[2 * Kk * Cc * 32 * 2];  // 40960 B
    __shared__ __align__(16) float sS[32 * KP];                          // 20480 B (scores, then bf16 A)
    __shared__ float sB9[(2 * Kk - 1) * (2 * Kk - 1)];

    __bf16* sK  = (__bf16*)uBuf;                           // [5][SPAN][64]
    __bf16* sQ  = (__bf16*)(uBuf + Kk * SPAN * Cc * 2);    // [32][64]
    __bf16* sVt = (__bf16*)uBuf;                           // [2][5][64][32]
    __bf16* sA  = (__bf16*)sS;                             // [32][KP] bf16

    const int tid = threadIdx.x;
    const int b  = blockIdx.x / (Hh * (Ww / 32));
    const int r0 = blockIdx.x % (Hh * (Ww / 32));
    const int h  = r0 / (Ww / 32);
    const int w0 = (r0 % (Ww / 32)) * 32;

    const int hs = iclamp(h - Kk / 2, 0, Hh - Kk);
    const int lo = iclamp(w0 - Kk / 2, 0, Ww - Kk);
    const int lane = tid & 31;
    const int wvid = tid >> 5;

    // ---- phase A staging: async global->LDS copies (no VGPR round-trip) ----
    for (int e = tid; e < Kk * SPAN * (Cc / 8); e += 256) {
        int r   = e / (SPAN * (Cc / 8));
        int rem = e % (SPAN * (Cc / 8));
        int cc  = rem / (Cc / 8);
        int c8  = (rem % (Cc / 8)) * 8;
        int gcol = iclamp(lo + cc, 0, Ww - 1);
        async_copy_b128(kk + (((size_t)b * Hh + hs + r) * Ww + gcol) * Cc + c8,
                        sK + (r * SPAN + cc) * Cc + c8);
    }
    for (int e = tid; e < 32 * (Cc / 8); e += 256) {
        int p  = e / (Cc / 8);
        int c8 = (e % (Cc / 8)) * 8;
        async_copy_b128(qq + (((size_t)b * Hh + h) * Ww + w0 + p) * Cc + c8,
                        sQ + p * Cc + c8);
    }
    if (tid < (2 * Kk - 1) * (2 * Kk - 1)) sB9[tid] = rpb[tid];
    __builtin_prefetch(vv + (((size_t)b * Hh + hs) * Ww + lo) * Cc, 0, 0);  // global_prefetch_b8
    async_wait0();
    __syncthreads();

    // ---- scores GEMM: 20 (mt,r,nt) tiles over 8 waves ----
#pragma unroll
    for (int i = 0; i < 3; ++i) {
        int t = wvid + i * 8;
        if (t < 2 * Kk * 2) {
            int mt  = t / (Kk * 2);
            int rr  = (t % (Kk * 2)) / 2;
            int nt  = t & 1;
            int lomt = iclamp(w0 + mt * 16 - Kk / 2, 0, Ww - Kk);
            int ccb  = (lomt - lo) + nt * 16 + (lane & 15);
            v8f acc = {};
#pragma unroll
            for (int kblk = 0; kblk < 2; ++kblk) {
                acc = __builtin_amdgcn_wmma_f32_16x16x32_bf16(
                    false, load_a_frag(sQ, mt * 16, kblk * 32, Cc, lane),
                    false, load_b_run(sK + (rr * SPAN + ccb) * Cc + kblk * 32 + ((lane >> 4) << 4)),
                    (short)0, acc, false, false);
            }
#pragma unroll
            for (int r = 0; r < 8; ++r) {
                int p = mt * 16 + r + ((lane >> 4) << 3);
                sS[p * KP + rr * 32 + nt * 16 + (lane & 15)] = acc[r];
            }
        }
    }
    __syncthreads();

    // ---- phase B staging: transposed V (per M-tile 32-col slabs) ----
    for (int e = tid; e < 2 * Kk * 8 * (Cc / 8); e += 256) {
        int chg  = e & 7;
        int colg = (e >> 3) & 7;
        int rr   = (e >> 6) % Kk;
        int mt   = e / (Kk * 64);
        int c8   = chg * 8;
        int lomt = iclamp(w0 + mt * 16 - Kk / 2, 0, Ww - Kk);
        bf8 col[4];
#pragma unroll
        for (int cj = 0; cj < 4; ++cj) {
            int gcol = iclamp(lomt + colg * 4 + cj, 0, Ww - 1);
            col[cj] = *(const bf8*)(vv + (((size_t)b * Hh + hs + rr) * Ww + gcol) * Cc + c8);
        }
#pragma unroll
        for (int j = 0; j < 8; ++j) {
            bf4 v4 = { col[0][j], col[1][j], col[2][j], col[3][j] };
            *(bf4*)(sVt + (((mt * Kk + rr) * Cc) + c8 + j) * 32 + colg * 4) = v4;
        }
    }

    // ---- per-pixel softmax (8 lanes per pixel, weights replicated in registers) ----
    const int p   = wvid * 4 + (lane >> 3);
    const int w   = w0 + p;
    const int mt  = p >> 4;
    const int lomt = iclamp(w0 + mt * 16 - Kk / 2, 0, Ww - Kk);
    const int ws_ = iclamp(w - Kk / 2, 0, Ww - Kk);
    const int off = ws_ - lomt;
    const int rh0 = hs  - h + (Kk - 1);
    const int rw0 = ws_ - w + (Kk - 1);

    float wgt[TAPS];
#pragma unroll
    for (int r = 0; r < Kk; ++r)
#pragma unroll
        for (int s = 0; s < Kk; ++s)
            wgt[r * Kk + s] = sS[p * KP + r * 32 + off + s]
                            + sB9[(rh0 + r) * (2 * Kk - 1) + (rw0 + s)];
    float mx = wgt[0];
#pragma unroll
    for (int t = 1; t < TAPS; ++t) mx = fmaxf(mx, wgt[t]);
    float sum = 0.f;
#pragma unroll
    for (int t = 0; t < TAPS; ++t) { wgt[t] = __expf(wgt[t] - mx); sum += wgt[t]; }
    float inv = 1.f / sum;
#pragma unroll
    for (int t = 0; t < TAPS; ++t) wgt[t] *= inv;
    __syncthreads();

    // ---- build sparse A matrix: zero-fill then scatter the 25 weights per pixel ----
    for (int e = tid; e < 32 * KP / 8; e += 256) ((bf8*)sA)[e] = (bf8)0;
    __syncthreads();
    if ((lane & 7) < Kk) {
        int r = lane & 7;
#pragma unroll
        for (int s = 0; s < Kk; ++s)
            sA[p * KP + r * 32 + off + s] = (__bf16)wgt[r * Kk + s];
    }
    __syncthreads();

    // ---- attn-V GEMM: 8 (mt,ntc) tiles over 8 waves, K = 160 in 5 steps ----
    {
        int gmt = wvid >> 2;
        int ntc = wvid & 3;
        v8f acc = {};
#pragma unroll
        for (int kblk = 0; kblk < Kk; ++kblk) {
            const __bf16* pa = sA + (gmt * 16 + (lane & 15)) * KP + kblk * 32 + ((lane >> 4) << 4);
            v16bf a = combine8(*(const bf8*)pa, *(const bf8*)(pa + 8));
            v16bf bf = load_b_run(
                sVt + ((gmt * Kk + kblk) * Cc + ntc * 16 + (lane & 15)) * 32 + ((lane >> 4) << 4));
            acc = __builtin_amdgcn_wmma_f32_16x16x32_bf16(
                false, a, false, bf, (short)0, acc, false, false);
        }
        int c = ntc * 16 + (lane & 15);
#pragma unroll
        for (int r = 0; r < 8; ++r) {
            int pix = gmt * 16 + r + ((lane >> 4) << 3);
            ctx[(((size_t)b * Hh + h) * Ww + w0 + pix) * Cc + c] = (__bf16)acc[r];
        }
    }
}

// ------------------------------------------------------------------
// Stage 3: output projection (bf16 WMMA) + gamma residual + transpose store
// ctx tile staged with async global->LDS copies.
// ------------------------------------------------------------------
__global__ __launch_bounds__(256) void natten_out_kernel(
    const __bf16* __restrict__ ctx, const float* __restrict__ Wp,
    const float* __restrict__ bp, const float* __restrict__ gamma,
    const float* __restrict__ kv, float* __restrict__ out)
{
    __shared__ __align__(16) __bf16 sX[Ww * Cc];
    __shared__ __align__(16) __bf16 sW[Cc * Cc];

    const int tid = threadIdx.x;
    const int b = blockIdx.x / Hh;
    const int h = blockIdx.x % Hh;

    const __bf16* src = ctx + (((size_t)b * Hh + h) * Ww) * Cc;
    for (int e = tid; e < Ww * Cc / 8; e += 256)
        async_copy_b128(src + e * 8, sX + e * 8);
    stage_weights_sw<Cc>(sW, Wp, tid);
    async_wait0();
    __syncthreads();

    const float g = gamma[0];
    const int lane = tid & 31;
    const int m0 = (tid >> 5) * 16;

#pragma unroll
    for (int nt = 0; nt < 4; ++nt) {
        v8f acc = {};
#pragma unroll
        for (int kblk = 0; kblk < 2; ++kblk) {
            acc = __builtin_amdgcn_wmma_f32_16x16x32_bf16(
                false, load_a_frag(sX, m0, kblk * 32, Cc, lane),
                false, load_b_frag_sw(sW, nt, kblk, lane),
                (short)0, acc, false, false);
        }
        int n = nt * 16 + (lane & 15);
        float bn = bp[n];
#pragma unroll
        for (int r = 0; r < 8; ++r) {
            int m = m0 + r + ((lane >> 4) << 3);
            size_t idx = (((size_t)b * Cc + n) * Hh + h) * Ww + m;
            out[idx] = g * (acc[r] + bn) + kv[idx];
        }
    }
}

// ------------------------------------------------------------------
extern "C" void kernel_launch(void* const* d_in, const int* in_sizes, int n_in,
                              void* d_out, int out_size, void* d_ws, size_t ws_size,
                              hipStream_t stream) {
    const float* q     = (const float*)d_in[0];
    const float* kv    = (const float*)d_in[1];
    const float* Wq    = (const float*)d_in[2];
    const float* bq    = (const float*)d_in[3];
    const float* Wkv   = (const float*)d_in[4];
    const float* bkv   = (const float*)d_in[5];
    const float* rpb   = (const float*)d_in[6];
    const float* Wp    = (const float*)d_in[7];
    const float* bp    = (const float*)d_in[8];
    const float* gamma = (const float*)d_in[9];
    float* out = (float*)d_out;

    const size_t NPC = (size_t)Bn * Hh * Ww * Cc;
    __bf16* qq  = (__bf16*)d_ws;
    __bf16* kk  = qq + NPC;
    __bf16* vv  = kk + NPC;
    __bf16* ctx = vv + NPC;

    natten_proj_kernel<<<Bn * Hh, 256, 0, stream>>>(q, kv, Wq, bq, Wkv, bkv, qq, kk, vv);
    natten_attn_kernel<<<Bn * Hh * (Ww / 32), 256, 0, stream>>>(qq, kk, vv, rpb, ctx);
    natten_out_kernel<<<Bn * Hh, 256, 0, stream>>>(ctx, Wp, bp, gamma, kv, out);
}
